// EncoderInput_15934328668405
// MI455X (gfx1250) — matrix operand order repaired and checked by reading, not compiled, not run on hardware
//
#include <hip/hip_runtime.h>
#include <hip/hip_bf16.h>
#include <stddef.h>
#include <stdint.h>

#define BB 128
#define TT 128
#define NN 128
#define MM 256

typedef __attribute__((ext_vector_type(16))) __bf16 v16bf;
typedef __attribute__((ext_vector_type(8)))  float  v8f;
typedef __attribute__((ext_vector_type(4)))  unsigned int v4u;
typedef __attribute__((ext_vector_type(8)))  int v8i;
typedef __attribute__((ext_vector_type(4)))  int v4i;

__device__ __forceinline__ float sigf(float x) { return 1.0f / (1.0f + __expf(-x)); }
__device__ __forceinline__ float ftanh(float x) {
  x = fminf(fmaxf(x, -15.f), 15.f);
  float e = __expf(2.f * x);
  return (e - 1.f) / (e + 1.f);
}

// A-fragment K offset inside a 32-wide chunk (ISA 7.12.2, 16-bit A 16x32).
__device__ __forceinline__ int a_klocal(int v, int half) {
  return ((v & 3) << 1) + ((v >> 2) << 4) + (half << 3);
}

__device__ __forceinline__ v8f wmma_bf16(v16bf a, v16bf b, v8f c) {
  return __builtin_amdgcn_wmma_f32_16x16x32_bf16(false, a, false, b, (short)0, c,
                                                 false, false);
}

// Fragment-major load from global: two global_load_b128 per fragment.
__device__ __forceinline__ v16bf load_frag(const __bf16* base, int tile, int lane) {
  const uint4* p = (const uint4*)(base + ((size_t)tile * 32 + lane) * 16);
  union { v16bf v; uint4 q[2]; } f;
  f.q[0] = p[0];
  f.q[1] = p[1];
  return f.v;
}

// Fragment-major load from LDS: two ds_load_b128 per fragment.
__device__ __forceinline__ v16bf load_frag_lds(const __bf16* base, int tile, int lane) {
  const uint4* p = (const uint4*)(base + ((size_t)tile * 32 + lane) * 16);
  union { v16bf v; uint4 q[2]; } f;
  f.q[0] = p[0];
  f.q[1] = p[1];
  return f.v;
}

// ---------------- prep: pack operands into WMMA fragment layout ----------------

__global__ void pack_b(const float* __restrict__ srcA, const float* __restrict__ srcB,
                       int split, int C, int nKK, int total, __bf16* __restrict__ dst) {
  int i = blockIdx.x * blockDim.x + threadIdx.x;
  if (i >= total) return;
  int e = i & 15;
  int lane = (i >> 4) & 31;
  int rest = i >> 9;
  int kk = rest % nKK;
  int ct = rest / nKK;
  int n = lane & 15, half = lane >> 4;
  int k = kk * 32 + (e & 7) + 8 * half + 16 * (e >> 3);
  int c = ct * 16 + n;
  float v = (k < split) ? srcA[(size_t)k * C + c] : srcB[(size_t)(k - split) * C + c];
  dst[i] = (__bf16)v;
}

__global__ void pack_a_xl(const float* __restrict__ X, __bf16* __restrict__ dst) {
  int i = blockIdx.x * blockDim.x + threadIdx.x;
  if (i >= BB * TT * NN) return;
  int e = i & 15;
  int lane = (i >> 4) & 31;
  int rest = i >> 9;
  int kk = rest & 3;  rest >>= 2;
  int bt = rest & 7;  int t = rest >> 3;
  int m = lane & 15, half = lane >> 4;
  int v = e >> 1;
  int klocal = a_klocal(v, half) + (e & 1);
  dst[i] = (__bf16)X[((size_t)(bt * 16 + m) * TT + t) * NN + kk * 32 + klocal];
}

__global__ void pack_a_xu(const float* __restrict__ X, __bf16* __restrict__ dst) {
  int i = blockIdx.x * blockDim.x + threadIdx.x;
  if (i >= BB * TT * NN) return;
  int e = i & 15;
  int lane = (i >> 4) & 31;
  int rest = i >> 9;
  int kk = rest & 3;  rest >>= 2;
  int nt = rest & 7;  int b = rest >> 5;
  int m = lane & 15, half = lane >> 4;
  int v = e >> 1;
  int klocal = a_klocal(v, half) + (e & 1);
  dst[i] = (__bf16)X[((size_t)b * TT + kk * 32 + klocal) * NN + nt * 16 + m];
}

__global__ void pack_a_h(const float* __restrict__ h0, __bf16* __restrict__ dst) {
  int i = blockIdx.x * blockDim.x + threadIdx.x;
  if (i >= BB * MM) return;
  int e = i & 15;
  int lane = (i >> 4) & 31;
  int rest = i >> 9;
  int kk = rest & 7;  int bt = rest >> 3;
  int m = lane & 15, half = lane >> 4;
  int v = e >> 1;
  int klocal = a_klocal(v, half) + (e & 1);
  dst[i] = (__bf16)h0[(size_t)(bt * 16 + m) * MM + kk * 32 + klocal];
}

__global__ void copy_f32(const float* __restrict__ s, float* __restrict__ d, int n) {
  int i = blockIdx.x * blockDim.x + threadIdx.x;
  if (i < n) d[i] = s[i];
}

// ---------------- UX = X^T @ Ud + bU (per batch), (B,N,T) ----------------
__global__ void ux_kernel(const __bf16* __restrict__ XU, const __bf16* __restrict__ UdS,
                          const float* __restrict__ bU, float* __restrict__ UX) {
  int ut = blockIdx.x;
  int nt = blockIdx.y;
  int b  = blockIdx.z;
  int lane = threadIdx.x;
  int half = lane >> 4;
  v8f acc = {};
#pragma unroll
  for (int kk = 0; kk < 4; ++kk) {
    v16bf a  = load_frag(XU, ((b * 8 + nt) * 4 + kk), lane);
    v16bf bf = load_frag(UdS, ut * 4 + kk, lane);
    acc = wmma_bf16(a, bf, acc);
  }
  int u = ut * 16 + (lane & 15);
  float bias = bU[u];
#pragma unroll
  for (int r = 0; r < 8; ++r) {
    int n = nt * 16 + r + 8 * half;
    UX[((size_t)b * NN + n) * TT + u] = acc[r] + bias;
  }
}

// ---------------- LSTM step: z = [x_t | h] @ [Wx;Wh] + b ----------------
// Weight fragment tiles for this block (4 gates x 12 K-chunks = 48KB) are
// DMA'd into LDS by the Tensor Data Mover as a 2D tile (4 rows x 12KB,
// row stride 192KB), then read back as ds_load_b128 fragments.
__global__ void lstm_step(const __bf16* __restrict__ XL, const __bf16* __restrict__ hin,
                          const __bf16* __restrict__ WcatS, const float* __restrict__ bias,
                          float* __restrict__ cf, __bf16* __restrict__ hswz,
                          __bf16* __restrict__ cswz, int t) {
  __shared__ __bf16 ldsW[4 * 12 * 32 * 16];   // [gate][kk][lane][16] = 48KB
  int bt = blockIdx.x;   // 0..7  batch tile
  int mt = blockIdx.y;   // 0..15 hidden tile
  int lane = threadIdx.x;
  int half = lane >> 4;

  // ---- TDM descriptor (D#) per CDNA5 ISA ----
  uint64_t gaddr = (uint64_t)(uintptr_t)WcatS + (uint64_t)mt * 12 * 1024;
  unsigned ldsoff = (unsigned)(uintptr_t)(void*)&ldsW[0];
  v4u g0;
  g0[0] = 1u;                                   // count=1, user mode
  g0[1] = ldsoff;                               // lds_addr (bytes)
  g0[2] = (unsigned)(gaddr & 0xffffffffu);      // global_addr[31:0]
  g0[3] = (unsigned)((gaddr >> 32) & 0x1ffffffu) | (2u << 30);  // addr[56:32], type=2
  v8i g1;
  g1[0] = (int)(2u << 16);                      // data_size = 4 bytes
  g1[1] = (int)(49152u << 16);                  // tensor_dim0[15:0] = 49152 (row elems)
  g1[2] = (int)(64u << 16);                     // dim0[31:16]=0 | tensor_dim1 = 64
  g1[3] = (int)(3072u << 16);                   // dim1 hi = 0 | tile_dim0 = 3072
  g1[4] = 4;                                    // tile_dim1 = 4, tile_dim2 = 0
  g1[5] = 49152;                                // tensor_dim0_stride[31:0] (4B units)
  g1[6] = 0;
  g1[7] = 0;
  v4i gz4 = {0, 0, 0, 0};
  v8i gz8 = {0, 0, 0, 0, 0, 0, 0, 0};
  __builtin_amdgcn_tensor_load_to_lds(g0, g1, gz4, gz4, gz8, 0);

  v8f acc0 = {}, acc1 = {}, acc2 = {}, acc3 = {};
  __builtin_amdgcn_s_wait_tensorcnt(0);
  for (int kk = 0; kk < 12; ++kk) {      // K = N + M = 384
    v16bf a = (kk < 4) ? load_frag(XL, ((t * 8 + bt) * 4 + kk), lane)
                       : load_frag(hin, bt * 8 + (kk - 4), lane);
    v16bf b0 = load_frag_lds(ldsW, 0 * 12 + kk, lane);
    acc0 = wmma_bf16(a, b0, acc0);
    v16bf b1 = load_frag_lds(ldsW, 1 * 12 + kk, lane);
    acc1 = wmma_bf16(a, b1, acc1);
    v16bf b2 = load_frag_lds(ldsW, 2 * 12 + kk, lane);
    acc2 = wmma_bf16(a, b2, acc2);
    v16bf b3 = load_frag_lds(ldsW, 3 * 12 + kk, lane);
    acc3 = wmma_bf16(a, b3, acc3);
  }
  int n = lane & 15;
  int mc = mt * 16 + n;
  float bi = bias[mc], bf_ = bias[MM + mc], bg = bias[2 * MM + mc],
        bo = bias[3 * MM + mc];
  int klocal = mc & 31;
  int half_d = (klocal >> 3) & 1;
  int v_d = ((klocal >> 4) << 2) + ((klocal & 7) >> 1);
  int e_d = 2 * v_d + (klocal & 1);
  int kk_d = mc >> 5;
#pragma unroll
  for (int r = 0; r < 8; ++r) {
    int m_a = r + 8 * half;
    int b = bt * 16 + m_a;
    float i = sigf(acc0[r] + bi);
    float f = sigf(acc1[r] + bf_);
    float g = ftanh(acc2[r] + bg);
    float o = sigf(acc3[r] + bo);
    size_t idx = (size_t)b * MM + mc;
    float c = f * cf[idx] + i * g;
    cf[idx] = c;
    float h = o * ftanh(c);
    size_t di = ((size_t)(bt * 8 + kk_d) * 32 + (m_a + 16 * half_d)) * 16 + e_d;
    hswz[di] = (__bf16)h;
    cswz[di] = (__bf16)c;
  }
}

// ---------------- attention part 1: w = [h;c] @ Wd + bW  (B x T) ----------------
__global__ void attn_w(const __bf16* __restrict__ hswz, const __bf16* __restrict__ cswz,
                       const __bf16* __restrict__ WdS, const float* __restrict__ bW,
                       float* __restrict__ w_buf) {
  int tid = threadIdx.x;
  int wave = tid >> 5, lane = tid & 31;
  int half = lane >> 4;
  int bt = blockIdx.x;
#pragma unroll
  for (int cti = 0; cti < 2; ++cti) {
    int ct = wave * 2 + cti;
    v8f acc = {};
#pragma unroll
    for (int kk = 0; kk < 16; ++kk) {
      v16bf a = (kk < 8) ? load_frag(hswz, bt * 8 + kk, lane)
                         : load_frag(cswz, bt * 8 + (kk - 8), lane);
      v16bf bf = load_frag(WdS, ct * 16 + kk, lane);
      acc = wmma_bf16(a, bf, acc);
    }
    int cc = ct * 16 + (lane & 15);
    float bw = bW[cc];
#pragma unroll
    for (int r = 0; r < 8; ++r)
      w_buf[(size_t)(bt * 16 + r + 8 * half) * TT + cc] = acc[r] + bw;
  }
}

// ---------------- attention part 2: e, softmax, scale (1 block per batch) -------
__global__ void attn_e(const float* __restrict__ X, const float* __restrict__ w_buf,
                       const float* __restrict__ UX, const float* __restrict__ vd,
                       const float* __restrict__ bv, float* __restrict__ out, int t) {
  __shared__ float w_s[TT];
  __shared__ float vd_s[TT];
  __shared__ float red[NN];
  int b = blockIdx.x;
  int n = threadIdx.x;          // 0..127
  w_s[n]  = w_buf[(size_t)b * TT + n];
  vd_s[n] = vd[n];
  __syncthreads();
  const float4* ux4 = (const float4*)(UX + ((size_t)b * NN + n) * TT);
  __builtin_prefetch(ux4, 0, 1);
  float acc = 0.f;
#pragma unroll 4
  for (int tq = 0; tq < TT / 4; ++tq) {
    float4 u = ux4[tq];
    acc += ftanh(w_s[4 * tq + 0] + u.x) * vd_s[4 * tq + 0];
    acc += ftanh(w_s[4 * tq + 1] + u.y) * vd_s[4 * tq + 1];
    acc += ftanh(w_s[4 * tq + 2] + u.z) * vd_s[4 * tq + 2];
    acc += ftanh(w_s[4 * tq + 3] + u.w) * vd_s[4 * tq + 3];
  }
  float e = acc + bv[0];
  // block softmax over n (128 values)
  red[n] = e;
  __syncthreads();
  for (int s = NN / 2; s > 0; s >>= 1) {
    if (n < s) red[n] = fmaxf(red[n], red[n + s]);
    __syncthreads();
  }
  float mx = red[0];
  __syncthreads();
  float ex = __expf(e - mx);
  red[n] = ex;
  __syncthreads();
  for (int s = NN / 2; s > 0; s >>= 1) {
    if (n < s) red[n] += red[n + s];
    __syncthreads();
  }
  float alpha = ex / red[0];
  size_t g = ((size_t)b * TT + t) * NN + n;
  out[g] = X[g] * alpha;
}

extern "C" void kernel_launch(void* const* d_in, const int* in_sizes, int n_in,
                              void* d_out, int out_size, void* d_ws, size_t ws_size,
                              hipStream_t stream) {
  const float* X  = (const float*)d_in[0];
  const float* h0 = (const float*)d_in[1];
  const float* s0 = (const float*)d_in[2];
  const float* Wx = (const float*)d_in[3];
  const float* Wh = (const float*)d_in[4];
  const float* bb = (const float*)d_in[5];
  const float* Wd = (const float*)d_in[6];
  const float* bW = (const float*)d_in[7];
  const float* Ud = (const float*)d_in[8];
  const float* bU = (const float*)d_in[9];
  const float* vd = (const float*)d_in[10];
  const float* bv = (const float*)d_in[11];
  float* out = (float*)d_out;

  char* ws = (char*)d_ws;
  size_t off = 0;
  auto alloc = [&](size_t bytes) -> void* {
    void* p = ws + off;
    off += (bytes + 255) & ~(size_t)255;
    return p;
  };
  float*  c_f32 = (float*)alloc((size_t)BB * MM * 4);
  float*  w_buf = (float*)alloc((size_t)BB * TT * 4);
  __bf16* hA    = (__bf16*)alloc((size_t)BB * MM * 2);
  __bf16* hB    = (__bf16*)alloc((size_t)BB * MM * 2);
  __bf16* cS    = (__bf16*)alloc((size_t)BB * MM * 2);
  __bf16* XL    = (__bf16*)alloc((size_t)BB * TT * NN * 2);
  __bf16* XU    = (__bf16*)alloc((size_t)BB * TT * NN * 2);
  __bf16* WcatS = (__bf16*)alloc((size_t)(NN + MM) * 4 * MM * 2);
  __bf16* WdS   = (__bf16*)alloc((size_t)2 * MM * TT * 2);
  __bf16* UdS   = (__bf16*)alloc((size_t)TT * TT * 2);
  float*  UX    = (float*)alloc((size_t)BB * NN * TT * 4);

  int nXL = BB * TT * NN;
  pack_a_xl<<<(nXL + 255) / 256, 256, 0, stream>>>(X, XL);
  pack_a_xu<<<(nXL + 255) / 256, 256, 0, stream>>>(X, XU);
  {
    int total = (NN + MM) * 4 * MM;
    pack_b<<<(total + 255) / 256, 256, 0, stream>>>(Wx, Wh, NN, 4 * MM, 12, total, WcatS);
  }
  {
    int total = 2 * MM * TT;
    pack_b<<<(total + 255) / 256, 256, 0, stream>>>(Wd, Wd, 2 * MM, TT, 16, total, WdS);
  }
  {
    int total = TT * TT;
    pack_b<<<(total + 255) / 256, 256, 0, stream>>>(Ud, Ud, TT, TT, 4, total, UdS);
  }
  pack_a_h<<<(BB * MM + 255) / 256, 256, 0, stream>>>(h0, hA);
  copy_f32<<<(BB * MM + 255) / 256, 256, 0, stream>>>(s0, c_f32, BB * MM);

  ux_kernel<<<dim3(TT / 16, NN / 16, BB), 32, 0, stream>>>(XU, UdS, bU, UX);

  __bf16* hin = hA;
  __bf16* hout = hB;
  for (int t = 0; t < TT; ++t) {
    lstm_step<<<dim3(BB / 16, MM / 16), 32, 0, stream>>>(XL, hin, WcatS, bb, c_f32,
                                                         hout, cS, t);
    attn_w<<<BB / 16, 128, 0, stream>>>(hout, cS, WdS, bW, w_buf);
    attn_e<<<BB, NN, 0, stream>>>(X, w_buf, UX, vd, bv, out, t);
    __bf16* tmp = hin; hin = hout; hout = tmp;
  }
}
